// MaskedSelfAttention_6210522710606
// MI455X (gfx1250) — matrix operand order repaired and checked by reading, not compiled, not run on hardware
//
#include <hip/hip_runtime.h>
#include <hip/hip_bf16.h>

#define T_SEQ 4096
#define D_IN  1024
#define H_DIM 64
#define B_N   4
#define WAVES 4

typedef __attribute__((ext_vector_type(16))) _Float16 v16h;
typedef __attribute__((ext_vector_type(8)))  _Float16 h8;
typedef __attribute__((ext_vector_type(8)))  float    v8f;

static __device__ __forceinline__ v8f wmma_f16(v16h a, v16h b, v8f c) {
    return __builtin_amdgcn_wmma_f32_16x16x32_f16(
        /*neg_a=*/false, a, /*neg_b=*/false, b,
        /*c_mod=*/(short)0, c, /*reuse_a=*/false, /*reuse_b=*/false);
}

// A-fragment (16x32 f16). Lane L holds row L%16; halves at K offsets
// (L>=16? 8:0)+0..7 and 16+(L>=16? 8:0)+0..7 (ISA 7.12.2).
// `p` points at (row base + k0) of a row-major f16 matrix.
static __device__ __forceinline__ v16h load_afrag(const _Float16* p, int hi) {
    h8 lo = *(const h8*)(p + hi * 8);
    h8 hh = *(const h8*)(p + 16 + hi * 8);
    v16h a;
#pragma unroll
    for (int i = 0; i < 8; ++i) { a[i] = lo[i]; a[8 + i] = hh[i]; }
    return a;
}

// B-fragment (32x16 f16). Lane L holds column N=L%16; K values
// (L>=16? 8:0)+0..7 and 16+(L>=16? 8:0)+0..7, matching the A-fragment.
// With the operand stored transposed (row-major N x K).
static __device__ __forceinline__ v16h load_bfrag(const _Float16* p, int hi) {
    h8 lo = *(const h8*)(p + hi * 8);
    h8 hh = *(const h8*)(p + 16 + hi * 8);
    v16h b;
#pragma unroll
    for (int i = 0; i < 8; ++i) { b[i] = lo[i]; b[8 + i] = hh[i]; }
    return b;
}

// ---------------------------------------------------------------------------
// Kernel 0: one-shot f32 -> f16 conversion of the three weight matrices.
// 64x1024 each; removes per-tile re-conversion from the projection kernel.
// ---------------------------------------------------------------------------
__global__ __launch_bounds__(256)
void cvt_weights(const float* __restrict__ Wq, const float* __restrict__ Wk,
                 const float* __restrict__ Wv, _Float16* __restrict__ Wqh,
                 _Float16* __restrict__ Wkh, _Float16* __restrict__ Wvh) {
    const int i = blockIdx.x * 256 + threadIdx.x;   // 65536 elements
    Wqh[i] = (_Float16)Wq[i];
    Wkh[i] = (_Float16)Wk[i];
    Wvh[i] = (_Float16)Wv[i];
}

// ---------------------------------------------------------------------------
// Kernel 1: Q = (xs Wq^T)*scale, K = xs Wk^T, V^T = (xs Wv^T)^T, f16 outputs.
// One wave computes a 16-row x 64-col tile of all three projections.
// Weights arrive as f16 (row-major N x K == ready-made B operand).
// ---------------------------------------------------------------------------
__global__ __launch_bounds__(WAVES * 32)
void proj_qkv(const float* __restrict__ xs, const _Float16* __restrict__ Wqh,
              const _Float16* __restrict__ Wkh, const _Float16* __restrict__ Wvh,
              _Float16* __restrict__ Qh, _Float16* __restrict__ Kh,
              _Float16* __restrict__ Vt) {
    const int lane = threadIdx.x & 31;
    const int wave = threadIdx.x >> 5;
    const int row  = lane & 15;
    const int hi   = lane >> 4;
    const long tile = (long)blockIdx.x * WAVES + wave;     // 0..1023
    const long m0   = tile * 16;                           // row in [0, B*T)

    v8f aq[4] = {}, ak[4] = {}, av[4] = {};
    const float* xrow = xs + (m0 + row) * D_IN;

    for (int k0 = 0; k0 < D_IN; k0 += 32) {
        v16h a;
#pragma unroll
        for (int i = 0; i < 8; ++i) {
            a[i]     = (_Float16)xrow[k0 + hi * 8 + i];
            a[8 + i] = (_Float16)xrow[k0 + 16 + hi * 8 + i];
        }
#pragma unroll
        for (int nb = 0; nb < 4; ++nb) {
            const size_t woff = (size_t)(nb * 16 + row) * D_IN + k0;
            v16h bq = load_bfrag(Wqh + woff, hi);
            v16h bk = load_bfrag(Wkh + woff, hi);
            v16h bv = load_bfrag(Wvh + woff, hi);
            aq[nb] = wmma_f16(a, bq, aq[nb]);
            ak[nb] = wmma_f16(a, bk, ak[nb]);
            av[nb] = wmma_f16(a, bv, av[nb]);
        }
    }

    const int b  = (int)(m0 / T_SEQ);
    const int t0 = (int)(m0 % T_SEQ);
    const float qscale = 0.125f;                 // fold 1/sqrt(64) into Q
#pragma unroll
    for (int nb = 0; nb < 4; ++nb) {
#pragma unroll
        for (int r = 0; r < 8; ++r) {
            const int m   = r + 8 * hi;          // C/D layout row
            const int col = nb * 16 + row;       // N = lane % 16
            Qh[(m0 + m) * H_DIM + col] = (_Float16)(aq[nb][r] * qscale);
            Kh[(m0 + m) * H_DIM + col] = (_Float16)ak[nb][r];
            Vt[((size_t)b * H_DIM + col) * T_SEQ + (t0 + m)] = (_Float16)av[nb][r];
        }
    }
}

// ---------------------------------------------------------------------------
// Kernel 2: causal flash attention. One wave owns 32 query rows (2 m-tiles);
// KV tiles of 32 keys. K/V fragments are loaded once per tile and reused by
// both m-tiles. Per-row softmax state lives in the VGPR slot that holds that
// row in the WMMA C/D layout; row reductions are shfl_xor over the half-wave.
// ---------------------------------------------------------------------------
__global__ __launch_bounds__(WAVES * 32)
void flash_attn(const _Float16* __restrict__ Qh, const _Float16* __restrict__ Kh,
                const _Float16* __restrict__ Vt, float* __restrict__ out) {
    __shared__ __align__(16) _Float16 plds[WAVES][2][16][32];

    const int lane = threadIdx.x & 31;
    const int wave = threadIdx.x >> 5;
    const int row  = lane & 15;
    const int hi   = lane >> 4;
    const int tile = blockIdx.x * WAVES + wave;            // 0..511
    const int b    = tile / (T_SEQ / 32);
    const int q0   = (tile % (T_SEQ / 32)) * 32;           // 32 rows, 32-aligned

    const _Float16* Qb = Qh + (size_t)b * T_SEQ * H_DIM;
    const _Float16* Kb = Kh + (size_t)b * T_SEQ * H_DIM;
    const _Float16* Vb = Vt + (size_t)b * H_DIM * T_SEQ;

    v16h qa[2][2];
#pragma unroll
    for (int mt = 0; mt < 2; ++mt) {
        const _Float16* qp = Qb + (size_t)(q0 + mt * 16 + row) * H_DIM;
        qa[mt][0] = load_afrag(qp, hi);                    // d 0..31
        qa[mt][1] = load_afrag(qp + 32, hi);               // d 32..63
    }

    float mi[2][8], li[2][8];
#pragma unroll
    for (int mt = 0; mt < 2; ++mt)
#pragma unroll
        for (int r = 0; r < 8; ++r) { mi[mt][r] = -3.4e38f; li[mt][r] = 0.0f; }
    v8f O[2][4] = {};

    for (int s0 = 0; s0 <= q0; s0 += 32) {                 // causal range
        // ---- K fragments for this 32-key tile (shared by both m-tiles) ----
        v16h kb[2][2];
#pragma unroll
        for (int nb = 0; nb < 2; ++nb) {
            const _Float16* kp = Kb + (size_t)(s0 + nb * 16 + row) * H_DIM;
            kb[nb][0] = load_bfrag(kp, hi);
            kb[nb][1] = load_bfrag(kp + 32, hi);
        }

        const bool diag = (s0 + 31 > q0);                  // only the last tile
#pragma unroll
        for (int mt = 0; mt < 2; ++mt) {
            // ---- S = (Q*scale) K^T : two 16x16 C blocks ----
            float sv[2][8];
#pragma unroll
            for (int nb = 0; nb < 2; ++nb) {
                v8f z = {};
                z = wmma_f16(qa[mt][0], kb[nb][0], z);
                z = wmma_f16(qa[mt][1], kb[nb][1], z);
#pragma unroll
                for (int r = 0; r < 8; ++r) {
                    float x = z[r];
                    if (diag) {
                        const int mg = q0 + mt * 16 + r + 8 * hi;
                        const int sg = s0 + nb * 16 + row;
                        if (sg > mg) x = -3.4e38f;
                    }
                    sv[nb][r] = x;
                }
            }

            // ---- online softmax (row = VGPR slot r + 8*hi) ----
#pragma unroll
            for (int r = 0; r < 8; ++r) {
                float mx = fmaxf(sv[0][r], sv[1][r]);
#pragma unroll
                for (int off = 1; off < 16; off <<= 1)
                    mx = fmaxf(mx, __shfl_xor(mx, off, 16));
                const float mnew  = fmaxf(mi[mt][r], mx);
                const float alpha = __expf(mi[mt][r] - mnew);
                const float p0 = __expf(sv[0][r] - mnew);
                const float p1 = __expf(sv[1][r] - mnew);
                sv[0][r] = p0; sv[1][r] = p1;
                float rs = p0 + p1;
#pragma unroll
                for (int off = 1; off < 16; off <<= 1)
                    rs += __shfl_xor(rs, off, 16);
                li[mt][r] = li[mt][r] * alpha + rs;
                mi[mt][r] = mnew;
#pragma unroll
                for (int nb = 0; nb < 4; ++nb) O[mt][nb][r] *= alpha;
            }

            // ---- P: C/D layout -> LDS (A-fragment staging) ----
#pragma unroll
            for (int nb = 0; nb < 2; ++nb)
#pragma unroll
                for (int r = 0; r < 8; ++r)
                    plds[wave][mt][r + 8 * hi][nb * 16 + row] = (_Float16)sv[nb][r];
        }

        // DS ops complete in-order within a wave; fence compiler + counter.
        __asm__ volatile("s_wait_dscnt 0" ::: "memory");
        v16h pa[2];
#pragma unroll
        for (int mt = 0; mt < 2; ++mt)
            pa[mt] = load_afrag(&plds[wave][mt][row][0], hi);

        // ---- O += P V : V^T fragments loaded once, used by both m-tiles ----
#pragma unroll
        for (int nb = 0; nb < 4; ++nb) {
            v16h vb = load_bfrag(Vb + (size_t)(nb * 16 + row) * T_SEQ + s0, hi);
            O[0][nb] = wmma_f16(pa[0], vb, O[0][nb]);
            O[1][nb] = wmma_f16(pa[1], vb, O[1][nb]);
        }
    }

    // ---- normalize + store ----
#pragma unroll
    for (int mt = 0; mt < 2; ++mt)
#pragma unroll
        for (int r = 0; r < 8; ++r) {
            const float inv = 1.0f / li[mt][r];
            const size_t orow =
                ((size_t)b * T_SEQ + q0 + mt * 16 + r + 8 * hi) * H_DIM;
#pragma unroll
            for (int nb = 0; nb < 4; ++nb)
                out[orow + nb * 16 + row] = O[mt][nb][r] * inv;
        }
}

extern "C" void kernel_launch(void* const* d_in, const int* in_sizes, int n_in,
                              void* d_out, int out_size, void* d_ws, size_t ws_size,
                              hipStream_t stream) {
    const float* xs = (const float*)d_in[0];
    const float* Wq = (const float*)d_in[1];
    const float* Wk = (const float*)d_in[2];
    const float* Wv = (const float*)d_in[3];

    const size_t elems = (size_t)B_N * T_SEQ * H_DIM;      // 1,048,576
    const size_t welems = (size_t)H_DIM * D_IN;            // 65,536
    _Float16* Qh  = (_Float16*)d_ws;                       // 2 MB
    _Float16* Kh  = Qh + elems;                            // 2 MB
    _Float16* Vt  = Kh + elems;                            // 2 MB
    _Float16* Wqh = Vt + elems;                            // 128 KB
    _Float16* Wkh = Wqh + welems;                          // 128 KB
    _Float16* Wvh = Wkh + welems;                          // 128 KB
    float* out = (float*)d_out;

    cvt_weights<<<dim3(welems / 256), dim3(256), 0, stream>>>(Wq, Wk, Wv,
                                                              Wqh, Wkh, Wvh);

    dim3 blk(WAVES * 32);
    proj_qkv<<<dim3(B_N * T_SEQ / 16 / WAVES), blk, 0, stream>>>(
        xs, Wqh, Wkh, Wvh, Qh, Kh, Vt);
    flash_attn<<<dim3(B_N * T_SEQ / 32 / WAVES), blk, 0, stream>>>(
        Qh, Kh, Vt, out);
}
